// VAREncoderCIFAR_17746804867716
// MI455X (gfx1250) — matrix operand order repaired and checked by reading, not compiled, not run on hardware
//
#include <hip/hip_runtime.h>
#include <hip/hip_bf16.h>

// ---------------------------------------------------------------------------
// Types for CDNA5 WMMA (wave32)
// ---------------------------------------------------------------------------
typedef __bf16 bf16_t;
typedef __attribute__((ext_vector_type(16))) __bf16 v16bf;
typedef __attribute__((ext_vector_type(8)))  __bf16 v8bf;
typedef __attribute__((ext_vector_type(8)))  float  v8f;
typedef int v4i __attribute__((vector_size(16)));

__device__ __forceinline__ bf16_t f2bf(float f) { return (bf16_t)f; }

// ---------------------------------------------------------------------------
// Optional gfx1250 async global->LDS copies (ASYNCcnt path)
// Probe-confirmed: builtin exists with 4 args; param0 is `v4i*` (generic).
// ---------------------------------------------------------------------------
#if defined(__has_builtin)
#  if __has_builtin(__builtin_amdgcn_global_load_async_to_lds_b128)
#    define HAVE_ASYNC_LDS 1
#  endif
#endif
#ifndef HAVE_ASYNC_LDS
#  define HAVE_ASYNC_LDS 0
#endif

#if HAVE_ASYNC_LDS
__device__ __forceinline__ void async_copy16(const void* g, void* l) {
  __builtin_amdgcn_global_load_async_to_lds_b128((v4i*)g, (v4i*)l, 0, 0);
}
__device__ __forceinline__ void async_wait0() {
  asm volatile("s_wait_asynccnt 0x0" ::: "memory");
}
#endif

// ---------------------------------------------------------------------------
// Problem constants
// ---------------------------------------------------------------------------
#define BATCH 512
#define NTOK  85
#define NPAD  96
#define DIM   512
#define HEADS 8
#define HD    64
#define ROWS  (BATCH * NTOK)       // 43520
#define DEPTH 6

// ---------------------------------------------------------------------------
// f32 -> bf16 transposing weight conversion: in [L][K][N] -> out [L][N][K]
// 32x32 LDS tile transpose, 256 threads (32x8)
// ---------------------------------------------------------------------------
__global__ __launch_bounds__(256)
void cvtT_kernel(const float* __restrict__ in, bf16_t* __restrict__ out,
                 int K, int N) {
  __shared__ float t[32][33];
  const int l = blockIdx.z;
  const float* src = in + (size_t)l * K * N;
  bf16_t* dst = out + (size_t)l * K * N;
  const int k0 = blockIdx.y * 32, n0 = blockIdx.x * 32;
  const int tx = threadIdx.x & 31, ty = threadIdx.x >> 5;
#pragma unroll
  for (int i = 0; i < 4; ++i)
    t[ty + 8 * i][tx] = src[(size_t)(k0 + ty + 8 * i) * N + n0 + tx];
  __syncthreads();
#pragma unroll
  for (int i = 0; i < 4; ++i)
    dst[(size_t)(n0 + ty + 8 * i) * K + k0 + tx] = f2bf(t[tx][ty + 8 * i]);
}

// ---------------------------------------------------------------------------
// LayerNorm: one 128-thread block per row of 512
// ---------------------------------------------------------------------------
template <bool F32OUT>
__global__ __launch_bounds__(128)
void ln_kernel(const float* __restrict__ x, const float* __restrict__ w,
               const float* __restrict__ b, bf16_t* __restrict__ outB,
               float* __restrict__ outF) {
  __shared__ float red[256];
  const int r = blockIdx.x, tid = threadIdx.x;
  const float* xr = x + (size_t)r * DIM;
  float v[4], s = 0.f, s2 = 0.f;
#pragma unroll
  for (int i = 0; i < 4; ++i) {
    v[i] = xr[tid + 128 * i];
    s += v[i];
    s2 += v[i] * v[i];
  }
  red[tid] = s;
  red[128 + tid] = s2;
  __syncthreads();
  for (int off = 64; off > 0; off >>= 1) {
    if (tid < off) {
      red[tid] += red[tid + off];
      red[128 + tid] += red[128 + tid + off];
    }
    __syncthreads();
  }
  float mu  = red[0] * (1.f / DIM);
  float var = red[128] * (1.f / DIM) - mu * mu;
  float inv = rsqrtf(var + 1e-5f);
#pragma unroll
  for (int i = 0; i < 4; ++i) {
    int c = tid + 128 * i;
    float y = (v[i] - mu) * inv * w[c] + b[c];
    if (F32OUT) outF[(size_t)r * DIM + c] = y;
    else        outB[(size_t)r * DIM + c] = f2bf(y);
  }
}

// ---------------------------------------------------------------------------
// bf16 WMMA GEMM with pre-transposed weights:
//   C[M,Nc] = A[M,K] @ Wt[Nc,K]^T  (+bias, epilogue)
//   256 threads (8 waves), block tile 128x128, wave tile 32x64, K-step 32.
//   mode 0: outF = acc + bias
//   mode 1: outF = acc + bias + res   (res may alias outF elementwise)
//   mode 2: outB = bf16(gelu(acc + bias))
// ---------------------------------------------------------------------------
#define TM 128
#define TN 128
#define TK 32

__global__ __launch_bounds__(256)
void gemm_bf16_kernel(const bf16_t* __restrict__ A, const bf16_t* __restrict__ Wt,
                      const float* __restrict__ bias, float* outF,
                      bf16_t* outB, const float* res,
                      int M, int Nc, int K, int mode) {
  __shared__ __align__(32) bf16_t As[TM * TK];   // [128 rows][32 k]
  __shared__ __align__(32) bf16_t Bs[TN * TK];   // [128 cols][32 k]

  const int tid  = threadIdx.x;
  const int wave = tid >> 5, lane = tid & 31;
  const int h = lane >> 4, ln = lane & 15;       // lane half / lane-in-half
  const int wm = wave & 3, wn = wave >> 2;       // wave grid 4(M) x 2(N)
  const int m0 = blockIdx.y * TM;
  const int n0 = blockIdx.x * TN;
  const int row = tid >> 1, col = (tid & 1) * 16;   // staging: 16 bf16 each

  v8f acc[2][4] = {};

  for (int k0 = 0; k0 < K; k0 += TK) {
    const bf16_t* ga = A  + (size_t)(m0 + row) * K + k0 + col;
    const bf16_t* gb = Wt + (size_t)(n0 + row) * K + k0 + col;
#if HAVE_ASYNC_LDS
    async_copy16(ga,     As + row * TK + col);
    async_copy16(ga + 8, As + row * TK + col + 8);
    async_copy16(gb,     Bs + row * TK + col);
    async_copy16(gb + 8, Bs + row * TK + col + 8);
#else
    ((v8bf*)(As + row * TK + col))[0] = ((const v8bf*)ga)[0];
    ((v8bf*)(As + row * TK + col))[1] = ((const v8bf*)ga)[1];
    ((v8bf*)(Bs + row * TK + col))[0] = ((const v8bf*)gb)[0];
    ((v8bf*)(Bs + row * TK + col))[1] = ((const v8bf*)gb)[1];
#endif
    if (k0 + TK < K) {
      __builtin_prefetch(ga + TK, 0, 1);
      __builtin_prefetch(gb + TK, 0, 1);
    }
#if HAVE_ASYNC_LDS
    async_wait0();
#endif
    __syncthreads();

    // fragments: contiguous k = 16*h + s in both A and B
    v16bf a[2], b[4];
#pragma unroll
    for (int mi = 0; mi < 2; ++mi) {
      int rr = wm * 32 + mi * 16 + ln;
      ((v8bf*)&a[mi])[0] = *(const v8bf*)(As + rr * TK + 16 * h);
      ((v8bf*)&a[mi])[1] = *(const v8bf*)(As + rr * TK + 16 * h + 8);
    }
#pragma unroll
    for (int nj = 0; nj < 4; ++nj) {
      int cc = wn * 64 + nj * 16 + ln;
      b[nj] = *(const v16bf*)(Bs + cc * TK + 16 * h);
    }
#pragma unroll
    for (int mi = 0; mi < 2; ++mi)
#pragma unroll
      for (int nj = 0; nj < 4; ++nj)
        acc[mi][nj] = __builtin_amdgcn_wmma_f32_16x16x32_bf16(
            false, a[mi], false, b[nj], (short)0, acc[mi][nj], false, false);
    __syncthreads();
  }

  // epilogue
#pragma unroll
  for (int mi = 0; mi < 2; ++mi)
#pragma unroll
    for (int nj = 0; nj < 4; ++nj) {
      int c = n0 + wn * 64 + nj * 16 + ln;
      float bv = bias[c];
#pragma unroll
      for (int r = 0; r < 8; ++r) {
        int rr = m0 + wm * 32 + mi * 16 + r + 8 * h;
        float v = acc[mi][nj][r] + bv;
        size_t o = (size_t)rr * Nc + c;
        if (mode == 0) {
          outF[o] = v;
        } else if (mode == 1) {
          outF[o] = v + res[o];
        } else {
          float g = 0.5f * v * (1.0f + erff(v * 0.70710678f));
          outB[o] = f2bf(g);
        }
      }
    }
}

// ---------------------------------------------------------------------------
// RoPE + repack: qkv f32 [ROWS,1536] -> q/k bf16 [B,H,96,64] (zero-padded)
//                and v transposed   -> vT bf16 [B,H,64,96]
// ---------------------------------------------------------------------------
__global__ __launch_bounds__(64)
void rope_pack_kernel(const float* __restrict__ qkv, const float* __restrict__ coords,
                      bf16_t* __restrict__ qp, bf16_t* __restrict__ kp,
                      bf16_t* __restrict__ vT) {
  const int blk = blockIdx.x;
  const int n  = blk % NPAD;
  const int bh = blk / NPAD;
  const int bb = bh >> 3, hh = bh & 7;
  const int d  = threadIdx.x;
  size_t qko = (size_t)bh * (NPAD * HD) + (size_t)n * HD + d;
  size_t vto = ((size_t)bh * HD + d) * NPAD + n;
  if (n >= NTOK) {
    qp[qko] = f2bf(0.f);
    kp[qko] = f2bf(0.f);
    vT[vto] = f2bf(0.f);
    return;
  }
  size_t rowb = (size_t)(bb * NTOK + n) * (3 * DIM);
  float qv = qkv[rowb +           hh * HD + d];
  float kv = qkv[rowb + DIM     + hh * HD + d];
  float vv = qkv[rowb + 2 * DIM + hh * HD + d];
  if (n >= 1) {
    int i = d & 15, g = d >> 4;
    float coord = coords[(n - 1) * 2 + (g >> 1)];
    float freq  = expf(-(float)i * 0.14391156672f);   // ln(10)/16
    float ang   = coord * freq * 6.28318530718f;
    float cs = cosf(ang), sn = sinf(ang);
    int pd = (g & 1) ? d - 16 : d + 16;
    float qpv = qkv[rowb +       hh * HD + pd];
    float kpv = qkv[rowb + DIM + hh * HD + pd];
    if ((g & 1) == 0) { qv = qv * cs - qpv * sn;  kv = kv * cs - kpv * sn; }
    else              { qv = qpv * sn + qv * cs;  kv = kpv * sn + kv * cs; }
  }
  qp[qko] = f2bf(qv);
  kp[qko] = f2bf(kv);
  vT[vto] = f2bf(vv);
}

// ---------------------------------------------------------------------------
// Attention: one 128-thread (4-wave) block per (batch, head).
// Static LDS: 24576 (q,k bf16) + 36864 (scores f32) = 61440 bytes.
// ---------------------------------------------------------------------------
__global__ __launch_bounds__(128)
void attn_kernel(const bf16_t* __restrict__ qp, const bf16_t* __restrict__ kp,
                 const bf16_t* __restrict__ vT, bf16_t* __restrict__ aout) {
  __shared__ __align__(32) unsigned char smem[24576 + NPAD * NPAD * 4];
  bf16_t* qs = (bf16_t*)smem;                 // [96][64]
  bf16_t* ks = qs + NPAD * HD;                // [96][64]
  float*  sc = (float*)(smem + 24576);        // [96][96]
  bf16_t* P  = (bf16_t*)smem;                 // [96][96], overlaps dead q/k

  const int bh = blockIdx.x;                  // b*8 + head
  const int tid = threadIdx.x;
  const int wave = tid >> 5, lane = tid & 31;
  const int h = lane >> 4, ln = lane & 15;
  const size_t base = (size_t)bh * NPAD * HD;

  // stage q, k (6144 bf16 each)
#pragma unroll
  for (int i = 0; i < 6; ++i) {
    int idx = (tid + i * 128) * 8;
#if HAVE_ASYNC_LDS
    async_copy16(qp + base + idx, qs + idx);
    async_copy16(kp + base + idx, ks + idx);
#else
    *(v8bf*)(qs + idx) = *(const v8bf*)(qp + base + idx);
    *(v8bf*)(ks + idx) = *(const v8bf*)(kp + base + idx);
#endif
  }
#if HAVE_ASYNC_LDS
  async_wait0();
#endif
  __syncthreads();

  // ---- scores: 6x6 tiles of 16x16, K = HD = 64 (2 chunks of 32) ----
  for (int t = wave; t < 36; t += 4) {
    int ti = t / 6, tj = t % 6;
    v8f acc = {};
#pragma unroll
    for (int c = 0; c < 2; ++c) {
      v16bf a, b;
      int row = ti * 16 + ln;
      ((v8bf*)&a)[0] = *(const v8bf*)(qs + row * HD + c * 32 + 16 * h);
      ((v8bf*)&a)[1] = *(const v8bf*)(qs + row * HD + c * 32 + 16 * h + 8);
      int tok = tj * 16 + ln;
      b = *(const v16bf*)(ks + tok * HD + c * 32 + 16 * h);
      acc = __builtin_amdgcn_wmma_f32_16x16x32_bf16(false, a, false, b,
                                                    (short)0, acc, false, false);
    }
#pragma unroll
    for (int r = 0; r < 8; ++r)
      sc[(ti * 16 + r + 8 * h) * NPAD + tj * 16 + ln] = acc[r];
  }
  __syncthreads();

  // ---- block mask + softmax: one thread per row ----
  if (tid < NPAD) {
    int qi = tid;
    if (qi >= NTOK) {
      for (int j = 0; j < NPAD; ++j) P[qi * NPAD + j] = f2bf(0.f);
    } else {
      const int qb = qi < 1 ? 0 : qi < 5 ? 1 : qi < 21 ? 2 : 3;
      const unsigned am[4] = {0x9u, 0x3u, 0x5u, 0x8u};  // M[i][j] -> bit j
      float mx = -3.0e38f;
      for (int j = 0; j < NTOK; ++j) {
        int kb = j < 1 ? 0 : j < 5 ? 1 : j < 21 ? 2 : 3;
        if ((am[qb] >> kb) & 1u) {
          float s = sc[qi * NPAD + j] * 0.125f;
          if (s > mx) mx = s;
        }
      }
      float sum = 0.f;
      for (int j = 0; j < NPAD; ++j) {
        float p = 0.f;
        if (j < NTOK) {
          int kb = j < 1 ? 0 : j < 5 ? 1 : j < 21 ? 2 : 3;
          if ((am[qb] >> kb) & 1u)
            p = expf(sc[qi * NPAD + j] * 0.125f - mx);
        }
        sc[qi * NPAD + j] = p;
        sum += p;
      }
      float inv = 1.0f / sum;
      for (int j = 0; j < NPAD; ++j)
        P[qi * NPAD + j] = f2bf(sc[qi * NPAD + j] * inv);
    }
  }
  __syncthreads();

  // ---- out = P @ V: 6x4 tiles, K = 96 tokens (3 chunks of 32) ----
  const int bb = bh >> 3, hh = bh & 7;
  for (int t = wave; t < 24; t += 4) {
    int ti = t >> 2, tj = t & 3;
    v8f acc = {};
#pragma unroll
    for (int c = 0; c < 3; ++c) {
      v16bf a, b;
      int row = ti * 16 + ln;
      ((v8bf*)&a)[0] = *(const v8bf*)(P + row * NPAD + c * 32 + 16 * h);
      ((v8bf*)&a)[1] = *(const v8bf*)(P + row * NPAD + c * 32 + 16 * h + 8);
      int d = tj * 16 + ln;
      b = *(const v16bf*)(vT + ((size_t)bh * HD + d) * NPAD + c * 32 + 16 * h);
      acc = __builtin_amdgcn_wmma_f32_16x16x32_bf16(false, a, false, b,
                                                    (short)0, acc, false, false);
    }
    int d = tj * 16 + ln;
#pragma unroll
    for (int r = 0; r < 8; ++r) {
      int n = ti * 16 + r + 8 * h;
      if (n < NTOK)
        aout[(size_t)(bb * NTOK + n) * DIM + hh * HD + d] = f2bf(acc[r]);
    }
  }
}

// ---------------------------------------------------------------------------
// Host-side orchestration
// ---------------------------------------------------------------------------
extern "C" void kernel_launch(void* const* d_in, const int* in_sizes, int n_in,
                              void* d_out, int out_size, void* d_ws, size_t ws_size,
                              hipStream_t stream) {
  (void)in_sizes; (void)n_in; (void)out_size; (void)ws_size;

  const float* x      = (const float*)d_in[0];
  const float* coords = (const float*)d_in[1];
  const float* ln1_w  = (const float*)d_in[2];
  const float* ln1_b  = (const float*)d_in[3];
  const float* qkv_w  = (const float*)d_in[4];
  const float* qkv_b  = (const float*)d_in[5];
  const float* proj_w = (const float*)d_in[6];
  const float* proj_b = (const float*)d_in[7];
  const float* ln2_w  = (const float*)d_in[8];
  const float* ln2_b  = (const float*)d_in[9];
  const float* fc1_w  = (const float*)d_in[10];
  const float* fc1_b  = (const float*)d_in[11];
  const float* fc2_w  = (const float*)d_in[12];
  const float* fc2_b  = (const float*)d_in[13];
  const float* lnf_w  = (const float*)d_in[14];
  const float* lnf_b  = (const float*)d_in[15];

  // workspace layout (256B aligned slabs)
  char* p = (char*)d_ws;
  auto alloc = [&](size_t bytes) -> char* {
    char* q = p;
    p += (bytes + 255) & ~(size_t)255;
    return q;
  };
  float*  xbuf = (float*)alloc((size_t)ROWS * DIM * 4);
  bf16_t* hbuf = (bf16_t*)alloc((size_t)ROWS * DIM * 2);
  float*  big  = (float*)alloc((size_t)ROWS * 3 * DIM * 4);   // qkv f32 / fc1 bf16
  bf16_t* qpk  = (bf16_t*)alloc((size_t)BATCH * HEADS * NPAD * HD * 2);
  bf16_t* kpk  = (bf16_t*)alloc((size_t)BATCH * HEADS * NPAD * HD * 2);
  bf16_t* vT   = (bf16_t*)alloc((size_t)BATCH * HEADS * HD * NPAD * 2);
  bf16_t* aout = (bf16_t*)alloc((size_t)ROWS * DIM * 2);
  bf16_t* wqkv = (bf16_t*)alloc((size_t)DEPTH * DIM * 3 * DIM * 2);   // [1536][512]
  bf16_t* wprj = (bf16_t*)alloc((size_t)DEPTH * DIM * DIM * 2);       // [512][512]
  bf16_t* wfc1 = (bf16_t*)alloc((size_t)DEPTH * DIM * 4 * DIM * 2);   // [2048][512]
  bf16_t* wfc2 = (bf16_t*)alloc((size_t)DEPTH * 4 * DIM * DIM * 2);   // [512][2048]

  // transposing weight conversion: W[L][K][N] -> Wt[L][N][K] in bf16
  cvtT_kernel<<<dim3(3 * DIM / 32, DIM / 32, DEPTH), 256, 0, stream>>>(
      qkv_w, wqkv, DIM, 3 * DIM);
  cvtT_kernel<<<dim3(DIM / 32, DIM / 32, DEPTH), 256, 0, stream>>>(
      proj_w, wprj, DIM, DIM);
  cvtT_kernel<<<dim3(4 * DIM / 32, DIM / 32, DEPTH), 256, 0, stream>>>(
      fc1_w, wfc1, DIM, 4 * DIM);
  cvtT_kernel<<<dim3(DIM / 32, 4 * DIM / 32, DEPTH), 256, 0, stream>>>(
      fc2_w, wfc2, 4 * DIM, DIM);

  (void)hipMemcpyAsync(xbuf, x, (size_t)ROWS * DIM * 4,
                       hipMemcpyDeviceToDevice, stream);

  const int GY = ROWS / TM;   // 340
  for (int l = 0; l < DEPTH; ++l) {
    // LN1 -> bf16
    ln_kernel<false><<<ROWS, 128, 0, stream>>>(xbuf, ln1_w + l * DIM, ln1_b + l * DIM,
                                               hbuf, nullptr);
    // QKV GEMM -> f32
    gemm_bf16_kernel<<<dim3(3 * DIM / TN, GY), 256, 0, stream>>>(
        hbuf, wqkv + (size_t)l * DIM * 3 * DIM, qkv_b + (size_t)l * 3 * DIM,
        big, nullptr, nullptr, ROWS, 3 * DIM, DIM, 0);
    // RoPE + pack q/k/vT (padded to 96 tokens)
    rope_pack_kernel<<<BATCH * HEADS * NPAD, 64, 0, stream>>>(big, coords, qpk, kpk, vT);
    // attention
    attn_kernel<<<BATCH * HEADS, 128, 0, stream>>>(qpk, kpk, vT, aout);
    // proj GEMM + residual -> xbuf
    gemm_bf16_kernel<<<dim3(DIM / TN, GY), 256, 0, stream>>>(
        aout, wprj + (size_t)l * DIM * DIM, proj_b + (size_t)l * DIM,
        xbuf, nullptr, xbuf, ROWS, DIM, DIM, 1);
    // LN2 -> bf16
    ln_kernel<false><<<ROWS, 128, 0, stream>>>(xbuf, ln2_w + l * DIM, ln2_b + l * DIM,
                                               hbuf, nullptr);
    // fc1 GEMM + gelu -> bf16 (reuse `big` slab)
    gemm_bf16_kernel<<<dim3(4 * DIM / TN, GY), 256, 0, stream>>>(
        hbuf, wfc1 + (size_t)l * DIM * 4 * DIM, fc1_b + (size_t)l * 4 * DIM,
        nullptr, (bf16_t*)big, nullptr, ROWS, 4 * DIM, DIM, 2);
    // fc2 GEMM + residual -> xbuf
    gemm_bf16_kernel<<<dim3(DIM / TN, GY), 256, 0, stream>>>(
        (bf16_t*)big, wfc2 + (size_t)l * 4 * DIM * DIM, fc2_b + (size_t)l * DIM,
        xbuf, nullptr, xbuf, ROWS, DIM, 4 * DIM, 1);
  }
  // final LN -> f32 output
  ln_kernel<true><<<ROWS, 128, 0, stream>>>(xbuf, lnf_w, lnf_b, nullptr, (float*)d_out);
}
